// CausalSelfAttention_14877766713804
// MI455X (gfx1250) — matrix operand order repaired and checked by reading, not compiled, not run on hardware
//
#include <hip/hip_runtime.h>
#include <hip/hip_bf16.h>

// ---------------------------------------------------------------------------
// CausalSelfAttention for MI455X (gfx1250): bf16 WMMA everywhere.
//   x(B,T,1024) @ qkv_w^T -> qkv ; RMSNorm+RoPE on q,k ; v = l0*v + l1*ve ;
//   causal flash-attention ; y @ c_proj_w^T + b -> out(B,T,1024)
// B=4 T=2048 dim=1024 H=8 hd=128
// ---------------------------------------------------------------------------

typedef __attribute__((ext_vector_type(16))) __bf16 v16bf;
typedef __attribute__((ext_vector_type(8)))  __bf16 v8bf;
typedef __attribute__((ext_vector_type(8)))  float  v8f;

__device__ __forceinline__ v16bf make_afrag(const __bf16* lo, const __bf16* hi8) {
    v8bf a = *(const v8bf*)lo;
    v8bf b = *(const v8bf*)hi8;
    v16bf r;
#pragma unroll
    for (int i = 0; i < 8; ++i) { r[i] = a[i]; r[8 + i] = b[i]; }
    return r;
}

// ---------------------------------------------------------------------------
// float -> bf16 elementwise convert
// ---------------------------------------------------------------------------
__global__ __launch_bounds__(256) void cvt_f32_bf16(const float* __restrict__ in,
                                                    __bf16* __restrict__ out, int n) {
    int i = blockIdx.x * 256 + threadIdx.x;
    if (i < n) out[i] = (__bf16)in[i];
}

// ---------------------------------------------------------------------------
// GEMM: C[M,N] = A[M,K] * W[N,K]^T   (A,W bf16 row-major, K%32==0, M%32==0, N%64==0)
// One wave computes a 32x64 tile (8 x 16x16 accumulators): 2 A-fragments,
// 4 B-fragments per 32-K block -> 8 WMMA, ~21 FLOP/byte from cache.
// WRITE_BF16=1 -> bf16 out ; else f32 out + bias.
// ---------------------------------------------------------------------------
template <int WRITE_BF16>
__global__ __launch_bounds__(256) void wmma_gemm(const __bf16* __restrict__ A,
                                                 const __bf16* __restrict__ W,
                                                 __bf16* __restrict__ outb,
                                                 float* __restrict__ outf,
                                                 const float* __restrict__ bias,
                                                 int M, int N, int K) {
    const int lane = threadIdx.x & 31;
    const int wave = threadIdx.x >> 5;
    const int hi   = lane >> 4;    // 0/1: selects K 8*hi..8*hi+7 and 16+8*hi..16+8*hi+7 of the 32-K block
    const int col  = lane & 15;    // N within 16 tile / M row for A frag
    const int nt64 = N >> 6;
    const int job  = blockIdx.x * 8 + wave;
    const int mt   = job / nt64;
    const int nb   = job % nt64;
    if (mt * 32 >= M) return;      // wave-uniform: EXEC stays all ones

    const __bf16* arow0 = A + (size_t)(mt * 32 + col) * K;
    const __bf16* arow1 = A + (size_t)(mt * 32 + 16 + col) * K;
    v8f acc[2][4] = {};

    for (int kb = 0; kb < K; kb += 32) {
        v16bf af0 = make_afrag(arow0 + kb + hi * 8, arow0 + kb + 16 + hi * 8);
        v16bf af1 = make_afrag(arow1 + kb + hi * 8, arow1 + kb + 16 + hi * 8);
        if (kb + 256 < K) {        // stream-ahead prefetch (global_prefetch_b8)
            __builtin_prefetch(arow0 + kb + 256, 0, 3);
            __builtin_prefetch(arow1 + kb + 256, 0, 3);
        }
#pragma unroll
        for (int c = 0; c < 4; ++c) {
            const __bf16* wrow = W + (size_t)(nb * 64 + c * 16 + col) * K + kb + hi * 8;
            v16bf bf = make_afrag(wrow, wrow + 16);
            if (kb + 256 < K) __builtin_prefetch(wrow + 256, 0, 3);
            acc[0][c] = __builtin_amdgcn_wmma_f32_16x16x32_bf16(false, af0, false, bf,
                                                                (short)0, acc[0][c], false, false);
            acc[1][c] = __builtin_amdgcn_wmma_f32_16x16x32_bf16(false, af1, false, bf,
                                                                (short)0, acc[1][c], false, false);
        }
    }
#pragma unroll
    for (int a = 0; a < 2; ++a) {
#pragma unroll
        for (int c = 0; c < 4; ++c) {
#pragma unroll
            for (int r = 0; r < 8; ++r) {
                int row = mt * 32 + a * 16 + r + 8 * hi;
                int n   = nb * 64 + c * 16 + col;
                float v = acc[a][c][r];
                if (WRITE_BF16) outb[(size_t)row * N + n] = (__bf16)v;
                else            outf[(size_t)row * N + n] = v + bias[n];
            }
        }
    }
}

// ---------------------------------------------------------------------------
// Per (b,t,h): RMSNorm(q,k), RoPE(q,k), v = l0*v + l1*ve.
// qkvb layout [B,T,3,H,128]; writes qh,kh [B,H,T,128] and vT [B,H,128,T] (bf16).
// blockDim = 128 (one thread per head-dim element), grid = B*T*H.
// ---------------------------------------------------------------------------
__global__ __launch_bounds__(128) void norm_rope_mix(const __bf16* __restrict__ qkvb,
                                                     const float* __restrict__ ve,
                                                     const float* __restrict__ lambdas,
                                                     __bf16* __restrict__ qh,
                                                     __bf16* __restrict__ kh,
                                                     __bf16* __restrict__ vT,
                                                     int B, int T, int H) {
    const int HD = 128;
    int idx = blockIdx.x;
    int h  = idx % H;
    int bt = idx / H;
    int t  = bt % T;
    int b  = bt / T;
    int d  = threadIdx.x;

    size_t base = (size_t)bt * 3072 + (size_t)h * HD;
    float q = (float)qkvb[base + 0 * 1024 + d];
    float k = (float)qkvb[base + 1 * 1024 + d];
    float v = (float)qkvb[base + 2 * 1024 + d];

    __shared__ float red[8];
    __shared__ float qs[128], ks[128];

    float sq = q * q, sk = k * k;
#pragma unroll
    for (int off = 16; off > 0; off >>= 1) {
        sq += __shfl_xor(sq, off, 32);
        sk += __shfl_xor(sk, off, 32);
    }
    int w = threadIdx.x >> 5;
    if ((threadIdx.x & 31) == 0) { red[w] = sq; red[4 + w] = sk; }
    __syncthreads();
    float msq = (red[0] + red[1] + red[2] + red[3]) * (1.0f / 128.0f);
    float msk = (red[4] + red[5] + red[6] + red[7]) * (1.0f / 128.0f);
    const float eps = 1.1920929e-7f;
    float qn = q * rsqrtf(msq + eps);
    float kn = k * rsqrtf(msk + eps);
    qs[d] = qn; ks[d] = kn;
    __syncthreads();

    // RoPE: half = 64 ; inv_freq[i] = 10000^(-i/64)
    int   i   = d & 63;
    float fi  = __expf(-9.210340372f * (float)i * (1.0f / 64.0f));
    float ang = (float)t * fi;
    float s, c;
    __sincosf(ang, &s, &c);
    float qr = (d < 64) ? (qs[d] * c - qs[d + 64] * s) : (qs[d - 64] * s + qs[d] * c);
    float kr = (d < 64) ? (ks[d] * c - ks[d + 64] * s) : (ks[d - 64] * s + ks[d] * c);

    float l0 = lambdas[0], l1 = lambdas[1];
    float vm = l0 * v + l1 * ve[(size_t)bt * 1024 + (size_t)h * HD + d];

    size_t bh = (size_t)(b * H + h);
    qh[(bh * T + t) * HD + d] = (__bf16)qr;
    kh[(bh * T + t) * HD + d] = (__bf16)kr;
    vT[(bh * HD + d) * T + t] = (__bf16)vm;
}

// ---------------------------------------------------------------------------
// Causal flash attention. Block = 256 thr = 8 waves; each wave owns 16 q rows.
// Per 32-key tile: 8 WMMA for S = Q K^T, online softmax, LDS round-trip to
// re-layout P into A-fragments, 8 WMMA for P V. ATTN_SCALE = 0.12.
// qh,kh [B,H,T,128]; vT [B,H,128,T]; yb out [B,T,H*128] bf16.
// ---------------------------------------------------------------------------
#define PST 40  // LDS P-tile row stride in halves (multiple of 8 -> 16B aligned)

__global__ __launch_bounds__(256) void flash_attn(const __bf16* __restrict__ qh,
                                                  const __bf16* __restrict__ kh,
                                                  const __bf16* __restrict__ vT,
                                                  __bf16* __restrict__ yb,
                                                  int B, int T, int H) {
    __shared__ __align__(16) __bf16 plds[8 * 16 * PST];
    const int HD   = 128;
    const int lane = threadIdx.x & 31;
    const int wave = threadIdx.x >> 5;
    const int hi   = lane >> 4;
    const int n    = lane & 15;
    const int bh   = blockIdx.y;
    const int h    = bh % H;
    const int b    = bh / H;
    const int q0   = blockIdx.x * 128 + wave * 16;

    const __bf16* Qb = qh + (size_t)bh * T * HD;
    const __bf16* Kb = kh + (size_t)bh * T * HD;
    const __bf16* Vb = vT + (size_t)bh * HD * T;

    // Q fragments (rows q0..q0+15, full 128 head dim = 4 K-blocks of 32)
    v16bf qa[4];
    {
        const __bf16* qrow = Qb + (size_t)(q0 + n) * HD;
#pragma unroll
        for (int db = 0; db < 4; ++db)
            qa[db] = make_afrag(qrow + db * 32 + hi * 8, qrow + db * 32 + 16 + hi * 8);
    }

    v8f acc[8] = {};
    float mrow[8], lrow[8];
#pragma unroll
    for (int r = 0; r < 8; ++r) { mrow[r] = -3.0e38f; lrow[r] = 0.0f; }

    __bf16* pl = plds + wave * 16 * PST;

    for (int s0 = 0; s0 < q0 + 16; s0 += 32) {
        // ---- S = Q K^T for 32 keys (two 16-key column tiles) ----
        v8f sc[2];
#pragma unroll
        for (int nt = 0; nt < 2; ++nt) {
            v8f cacc = {};
            const __bf16* krow = Kb + (size_t)(s0 + nt * 16 + n) * HD;
            __builtin_prefetch(krow + 32 * HD, 0, 3);   // next key tile
#pragma unroll
            for (int db = 0; db < 4; ++db) {
                v16bf bf = make_afrag(krow + db * 32 + hi * 8, krow + db * 32 + 16 + hi * 8);
                cacc = __builtin_amdgcn_wmma_f32_16x16x32_bf16(false, qa[db], false, bf,
                                                               (short)0, cacc, false, false);
            }
            sc[nt] = cacc;
        }

        // ---- scale, causal mask, online softmax ----
        float newm[8];
#pragma unroll
        for (int r = 0; r < 8; ++r) {
            int row = q0 + r + 8 * hi;
            float v0 = sc[0][r] * 0.12f;
            float v1 = sc[1][r] * 0.12f;
            if (s0 + n > row)      v0 = -3.0e38f;
            if (s0 + 16 + n > row) v1 = -3.0e38f;
            sc[0][r] = v0; sc[1][r] = v1;
            float t = fmaxf(v0, v1);
#pragma unroll
            for (int off = 1; off < 16; off <<= 1) t = fmaxf(t, __shfl_xor(t, off, 32));
            newm[r] = fmaxf(mrow[r], t);
        }
#pragma unroll
        for (int r = 0; r < 8; ++r) {
            float p0    = __expf(sc[0][r] - newm[r]);
            float p1    = __expf(sc[1][r] - newm[r]);
            float alpha = __expf(mrow[r] - newm[r]);
            float rs = p0 + p1;
#pragma unroll
            for (int off = 1; off < 16; off <<= 1) rs += __shfl_xor(rs, off, 32);
            lrow[r] = lrow[r] * alpha + rs;
            mrow[r] = newm[r];
            sc[0][r] = p0; sc[1][r] = p1;
#pragma unroll
            for (int dv = 0; dv < 8; ++dv) acc[dv][r] *= alpha;
        }

        // ---- P: C-layout -> A-fragment layout via per-wave LDS tile ----
#pragma unroll
        for (int r = 0; r < 8; ++r) {
            int row = r + 8 * hi;
            pl[row * PST + n]      = (__bf16)sc[0][r];
            pl[row * PST + 16 + n] = (__bf16)sc[1][r];
        }
        asm volatile("s_wait_dscnt 0" ::: "memory");
        v16bf pa = make_afrag(pl + n * PST + hi * 8, pl + n * PST + 16 + hi * 8);

        // ---- y += P V ----
#pragma unroll
        for (int dv = 0; dv < 8; ++dv) {
            const __bf16* vrow = Vb + (size_t)(dv * 16 + n) * T + s0 + hi * 8;
            v16bf vf = make_afrag(vrow, vrow + 16);
            __builtin_prefetch(vrow + 32, 0, 3);        // next key tile of V
            acc[dv] = __builtin_amdgcn_wmma_f32_16x16x32_bf16(false, pa, false, vf,
                                                              (short)0, acc[dv], false, false);
        }
    }

    // ---- epilogue: y /= l ; write [B,T,H*128] bf16 ----
#pragma unroll
    for (int r = 0; r < 8; ++r) {
        float inv = 1.0f / lrow[r];
        int trow  = q0 + r + 8 * hi;
        __bf16* yrow = yb + ((size_t)(b * T + trow)) * 1024 + (size_t)h * HD;
#pragma unroll
        for (int dv = 0; dv < 8; ++dv)
            yrow[dv * 16 + n] = (__bf16)(acc[dv][r] * inv);
    }
}

// ---------------------------------------------------------------------------
extern "C" void kernel_launch(void* const* d_in, const int* in_sizes, int n_in,
                              void* d_out, int out_size, void* d_ws, size_t ws_size,
                              hipStream_t stream) {
    const float* x       = (const float*)d_in[0];
    const float* ve      = (const float*)d_in[1];
    const float* qkv_w   = (const float*)d_in[2];
    const float* lambdas = (const float*)d_in[3];
    const float* cpw     = (const float*)d_in[4];
    const float* cpb     = (const float*)d_in[5];
    float* out = (float*)d_out;

    const int B = 4, T = 2048, DIM = 1024, H = 8, HD = 128;
    const int M = B * T;             // 8192
    const int N_QKV = 3 * H * HD;    // 3072

    char* ws = (char*)d_ws;
    size_t off = 0;
    auto carve = [&](size_t bytes) -> void* {
        void* p = (void*)(ws + off);
        off += (bytes + 255) & ~(size_t)255;
        return p;
    };
    __bf16* xb    = (__bf16*)carve((size_t)M * DIM * 2);
    __bf16* wqkv  = (__bf16*)carve((size_t)N_QKV * DIM * 2);
    __bf16* wproj = (__bf16*)carve((size_t)DIM * DIM * 2);
    __bf16* qkvb  = (__bf16*)carve((size_t)M * N_QKV * 2);
    __bf16* qhb   = (__bf16*)carve((size_t)B * H * T * HD * 2);
    __bf16* khb   = (__bf16*)carve((size_t)B * H * T * HD * 2);
    __bf16* vTb   = (__bf16*)carve((size_t)B * H * HD * T * 2);
    __bf16* yb    = (__bf16*)carve((size_t)M * DIM * 2);

    // 1) convert inputs/weights to bf16
    {
        int n1 = M * DIM;
        cvt_f32_bf16<<<(n1 + 255) / 256, 256, 0, stream>>>(x, xb, n1);
        int n2 = N_QKV * DIM;
        cvt_f32_bf16<<<(n2 + 255) / 256, 256, 0, stream>>>(qkv_w, wqkv, n2);
        int n3 = DIM * DIM;
        cvt_f32_bf16<<<(n3 + 255) / 256, 256, 0, stream>>>(cpw, wproj, n3);
    }
    // 2) QKV projection: [8192,1024] x [3072,1024]^T -> bf16 [B,T,3,H,128]
    {
        int jobs = (M / 32) * (N_QKV / 64);   // 12288 waves
        wmma_gemm<1><<<jobs / 8, 256, 0, stream>>>(xb, wqkv, qkvb, nullptr, nullptr,
                                                   M, N_QKV, DIM);
    }
    // 3) RMSNorm + RoPE + value mix
    norm_rope_mix<<<B * T * H, 128, 0, stream>>>(qkvb, ve, lambdas, qhb, khb, vTb, B, T, H);
    // 4) causal flash attention
    {
        dim3 grid(T / 128, B * H);
        flash_attn<<<grid, 256, 0, stream>>>(qhb, khb, vTb, yb, B, T, H);
    }
    // 5) output projection: [8192,1024] x [1024,1024]^T + bias -> f32 out
    {
        int jobs = (M / 32) * (DIM / 64);     // 4096 waves
        wmma_gemm<0><<<jobs / 8, 256, 0, stream>>>(yb, wproj, nullptr, out, cpb,
                                                   M, DIM, DIM);
    }
}
